// BertSelfAttention_24687472017696
// MI455X (gfx1250) — matrix operand order repaired and checked by reading, not compiled, not run on hardware
//
#include <hip/hip_runtime.h>

typedef _Float16 v8h  __attribute__((ext_vector_type(8)));
typedef _Float16 v16h __attribute__((ext_vector_type(16)));
typedef float    v8f  __attribute__((ext_vector_type(8)));
typedef unsigned int u32x4 __attribute__((ext_vector_type(4)));
typedef int          i32x8 __attribute__((ext_vector_type(8)));
typedef int          i32x4 __attribute__((ext_vector_type(4)));

#define NUM_HEADS 16
#define HIDDEN    1024
#define HEAD_DIM  64
#define BATCH     4
#define SEQ       2048

// ---- WMMA fragment helpers (layouts per CDNA5 ISA 7.12.2, wave32) ----------
// A (16x32 f16): lane holds row m = lane%16; elems 0-7 = K[g*8 .. g*8+7],
//                elems 8-15 = K[16+g*8 .. 16+g*8+7], g = lane/16.
// B (32x16 f16): lane holds col n = lane%16; elems 0-7 = K[g*8 .. g*8+7],
//                elems 8-15 = K[16+g*8 .. 16+g*8+7], g = lane/16.
// C (16x16 f32): VGPR r, lane l -> (m = 8*(l/16)+r, n = l%16).

__device__ __forceinline__ v16h combine8(v8h lo, v8h hi) {
  v16h r;
#pragma unroll
  for (int i = 0; i < 8; ++i) { r[i] = lo[i]; r[i + 8] = hi[i]; }
  return r;
}

__device__ __forceinline__ v16h load_afrag(const _Float16* row, int g) {
  v8h lo = *(const v8h*)(row + g * 8);
  v8h hi = *(const v8h*)(row + 16 + g * 8);
  return combine8(lo, hi);
}

__device__ __forceinline__ v16h load_bfrag(const _Float16* col, int g) {
  v8h lo = *(const v8h*)(col + g * 8);
  v8h hi = *(const v8h*)(col + 16 + g * 8);
  return combine8(lo, hi);
}

__device__ __forceinline__ v8f wmma_f16(v16h a, v16h b, v8f c) {
  return __builtin_amdgcn_wmma_f32_16x16x32_f16(false, a, false, b, (short)0, c,
                                                false, false);
}

// ---- CDNA5 async global->LDS copy (16B per lane), tracked by ASYNCcnt ------
__device__ __forceinline__ void async_copy_b128(void* lds, const void* gaddr) {
  unsigned lds_off = (unsigned)(uintptr_t)lds;  // low 32 bits = LDS byte addr
  asm volatile("global_load_async_to_lds_b128 %0, %1, off"
               :
               : "v"(lds_off), "v"(gaddr)
               : "memory");
}

__device__ __forceinline__ void wait_async0() {
  asm volatile("s_wait_asynccnt 0" ::: "memory");
}

// ---- CDNA5 Tensor Data Mover: 2-D tile load, HW-applied LDS row padding ----
// D# packing per ISA 8.3/8.4.  2-byte elements; pad_interval/pad_amount in the
// ISA encodings (interval: 2^(v+1) DWORDs, amount: v+1 DWORDs).
// This toolchain's builtin takes 6 args (clang-23 / therock-10.0 form):
//   (u32x4 g0, i32x8 g1, i32x4 g2, i32x4 g3, i32x8 extra, i32 cpol)
__device__ __forceinline__ void tdm_load_2d(unsigned lds_addr, const void* gptr,
                                            unsigned tensor_d0,
                                            unsigned tensor_d1,
                                            unsigned tile_d0, unsigned tile_d1,
                                            unsigned stride_d0,
                                            unsigned pad_interval,
                                            unsigned pad_amount) {
  unsigned long long ga = (unsigned long long)(uintptr_t)gptr;
  u32x4 g0;
  g0[0] = 1u;                                   // count=1, user descriptor
  g0[1] = lds_addr;                             // LDS byte address
  g0[2] = (unsigned)ga;                         // global addr [31:0]
  g0[3] = (unsigned)((ga >> 32) & 0x01FFFFFFu)  // global addr [56:32]
          | (2u << 30);                         // type = 2 ("image")
  i32x8 g1;
  g1[0] = (int)((1u << 16)                      // data_size = 1 -> 2 bytes
                | (1u << 20)                    // pad_enable
                | (pad_interval << 22) | (pad_amount << 25));
  g1[1] = (int)((tensor_d0 & 0xFFFFu) << 16);                     // dim0 lo16
  g1[2] = (int)((tensor_d0 >> 16) | ((tensor_d1 & 0xFFFFu) << 16));
  g1[3] = (int)((tensor_d1 >> 16) | (tile_d0 << 16));             // tile_dim0
  g1[4] = (int)tile_d1;                         // tile_dim1 (tile_dim2 = 0)
  g1[5] = (int)stride_d0;                       // tensor_dim0_stride lo32
  g1[6] = 0;
  g1[7] = 0;
  i32x4 z4 = {0, 0, 0, 0};                      // groups 2/3 unused (2-D)
  i32x8 z8 = {0, 0, 0, 0, 0, 0, 0, 0};
  __builtin_amdgcn_tensor_load_to_lds(g0, g1, z4, z4, z8, 0);
}

__device__ __forceinline__ void wait_tensor0() {
  __builtin_amdgcn_s_wait_tensorcnt((short)0);
}

// ============================================================================
// Kernel 0: elementwise fp32 -> f16 conversion (HBM-bound pre-pass)
// ============================================================================
__global__ __launch_bounds__(256) void f32_to_f16_kernel(
    const float* __restrict__ src, _Float16* __restrict__ dst, int n) {
  int i = (blockIdx.x * 256 + threadIdx.x) * 8;
  if (i < n) {
    float4 a = *(const float4*)(src + i);
    float4 b = *(const float4*)(src + i + 4);
    v8h o;
    o[0] = (_Float16)a.x; o[1] = (_Float16)a.y;
    o[2] = (_Float16)a.z; o[3] = (_Float16)a.w;
    o[4] = (_Float16)b.x; o[5] = (_Float16)b.y;
    o[6] = (_Float16)b.z; o[7] = (_Float16)b.w;
    *(v8h*)(dst + i) = o;
  }
}

// ============================================================================
// Kernel 1: fused QKV projection  Y = Xh @ Wh^T + b   (z selects q/k/v)
//   Block tile 128(M) x 128(N), K-chunk 32, double-buffered LDS staging via
//   the Tensor Data Mover (1 descriptor per tile; HW pads rows 64B + 16B).
//   8 waves each compute a 32x64 subtile (8 WMMAs / chunk).
// ============================================================================
__global__ __launch_bounds__(256) void qkv_proj_kernel(
    const _Float16* __restrict__ Xh, const _Float16* __restrict__ Wh,
    const float* __restrict__ bq, const float* __restrict__ bk,
    const float* __restrict__ bv, _Float16* __restrict__ ws) {
  constexpr int KPAD = 40;  // 32 f16 + 8 pad = 80B row stride (TDM-padded)
  __shared__ _Float16 Xs[2][128 * KPAD];
  __shared__ _Float16 Wt[2][128 * KPAD];

  const int z = blockIdx.z;
  const float* bias = (z == 0) ? bq : (z == 1) ? bk : bv;
  const _Float16* W = Wh + (size_t)z * (HIDDEN * HIDDEN);
  _Float16* out = ws + (size_t)z * ((size_t)BATCH * NUM_HEADS * SEQ * HEAD_DIM);

  const int m0   = blockIdx.x * 128;
  const int n0   = blockIdx.y * 128;
  const int tid  = threadIdx.x;
  const int lane = tid & 31;
  const int wave = tid >> 5;
  const int g    = lane >> 4;
  const int ln   = lane & 15;
  const int wm   = wave & 3;   // M slot (x32 rows)
  const int wn   = wave >> 2;  // N slot (x64 cols)

  // one TDM descriptor per tile; X from wave 0's queue, W from wave 1's
  auto stage = [&](int buf, int kk) {
    if (wave == 0) {
      tdm_load_2d((unsigned)(uintptr_t)&Xs[buf][0],
                  Xh + (size_t)m0 * HIDDEN + kk,
                  /*tensor_d0=*/HIDDEN, /*tensor_d1=*/BATCH * SEQ,
                  /*tile_d0=*/32, /*tile_d1=*/128, /*stride_d0=*/HIDDEN,
                  /*pad_interval=*/3 /*16 DW = 64B*/, /*pad_amount=*/3 /*4 DW*/);
    } else if (wave == 1) {
      tdm_load_2d((unsigned)(uintptr_t)&Wt[buf][0],
                  W + (size_t)n0 * HIDDEN + kk,
                  /*tensor_d0=*/HIDDEN, /*tensor_d1=*/HIDDEN,
                  /*tile_d0=*/32, /*tile_d1=*/128, /*stride_d0=*/HIDDEN,
                  /*pad_interval=*/3, /*pad_amount=*/3);
    }
  };

  v8f acc[2][4];
#pragma unroll
  for (int i = 0; i < 2; ++i)
#pragma unroll
    for (int j = 0; j < 4; ++j) acc[i][j] = v8f{};

  stage(0, 0);
  wait_tensor0();      // all waves: no-op unless this wave issued TDM
  __syncthreads();

  int cur = 0;
  for (int kk = 0; kk < HIDDEN; kk += 32) {
    if (kk + 32 < HIDDEN) stage(cur ^ 1, kk + 32);  // overlap DMA with WMMA

    v16h a[2], bfr[4];
#pragma unroll
    for (int i = 0; i < 2; ++i)
      a[i] = load_afrag(&Xs[cur][(wm * 32 + i * 16 + ln) * KPAD], g);
#pragma unroll
    for (int j = 0; j < 4; ++j)
      bfr[j] = load_bfrag(&Wt[cur][(wn * 64 + j * 16 + ln) * KPAD], g);
#pragma unroll
    for (int i = 0; i < 2; ++i)
#pragma unroll
      for (int j = 0; j < 4; ++j) acc[i][j] = wmma_f16(a[i], bfr[j], acc[i][j]);

    wait_tensor0();
    __syncthreads();
    cur ^= 1;
  }

  // epilogue: +bias, f16, scatter into [b][h][s][d]
#pragma unroll
  for (int j = 0; j < 4; ++j) {
    int n = n0 + wn * 64 + j * 16 + ln;
    float bj = bias[n];
    int head = n >> 6, d = n & 63;
#pragma unroll
    for (int i = 0; i < 2; ++i) {
#pragma unroll
      for (int r = 0; r < 8; ++r) {
        int mg = m0 + wm * 32 + i * 16 + 8 * g + r;
        int bb = mg >> 11, s = mg & (SEQ - 1);
        out[(size_t)((bb * NUM_HEADS + head) * SEQ + s) * HEAD_DIM + d] =
            (_Float16)(acc[i][j][r] + bj);
      }
    }
  }
}

// ============================================================================
// Kernel 2: flash attention, double-buffered 32-key tiles.
//   K tile staged via per-lane async global->LDS copies (ASYNCcnt path);
//   V tile (needs transpose) prefetched to registers one iteration ahead.
// ============================================================================
__global__ __launch_bounds__(256) void attention_kernel(
    const _Float16* __restrict__ Q, const _Float16* __restrict__ K,
    const _Float16* __restrict__ V, const float* __restrict__ mask,
    float* __restrict__ out) {
  __shared__ _Float16 Kt[2][32 * 72];     // [n][d], stride 72
  __shared__ _Float16 Vt[2][64 * 40];     // transposed [d][n], stride 40
  __shared__ _Float16 Pb[8][16 * 40];     // per-wave P tile [m][n], stride 40

  const int bh = blockIdx.x;              // b*16 + h
  const int b  = bh >> 4, h = bh & 15;
  const int s0 = blockIdx.y * 128;
  const int tid = threadIdx.x, lane = tid & 31, wave = tid >> 5;
  const int g = lane >> 4, ln = lane & 15;

  const int vrow = tid >> 3;              // V staging coords
  const int vcs  = (tid & 7) * 8;

  auto stage_k = [&](int buf, int nb) {   // 32 rows x 32 f16 (d 0..31) = 128 b128 chunks
    if (tid < 128) {
      int row = tid >> 2, off = (tid & 3) * 8;
      async_copy_b128(&Kt[buf][row * 72 + off],
                      K + (size_t)(bh * SEQ + nb + row) * HEAD_DIM + off);
    }
  };
  auto load_v = [&](int nb) -> v8h {
    return *(const v8h*)(V + (size_t)(bh * SEQ + nb + vrow) * HEAD_DIM + vcs);
  };
  auto store_v = [&](int buf, v8h v) {    // transpose to [d][n]
#pragma unroll
    for (int i = 0; i < 8; ++i) Vt[buf][(vcs + i) * 40 + vrow] = v[i];
  };

  // Q fragments: loaded once, straight from global in A-fragment layout
  const _Float16* qrow =
      Q + (size_t)((bh * SEQ) + s0 + wave * 16 + ln) * HEAD_DIM;
  const v16h qa0 = load_afrag(qrow, g);
  const v16h qa1 = load_afrag(qrow + 32, g);

  v8f o[4];
#pragma unroll
  for (int dt = 0; dt < 4; ++dt) o[dt] = v8f{};
  float mrun[8], lrun[8];
#pragma unroll
  for (int r = 0; r < 8; ++r) { mrun[r] = -1e30f; lrun[r] = 0.0f; }

  // prologue: stage tile 0
  stage_k(0, 0);
  store_v(0, load_v(0));
  wait_async0();
  __syncthreads();

  int cur = 0;
  for (int nb = 0; nb < SEQ; nb += 32) {
    const bool more = (nb + 32) < SEQ;
    v8h vnext;
    if (more) {                       // issue next tile's loads early
      stage_k(cur ^ 1, nb + 32);
      vnext = load_v(nb + 32);
    }

    // scores: 16(m) x 32(n) per wave; K=64 in two 32-chunks
    v8f sc[2];
#pragma unroll
    for (int j = 0; j < 2; ++j) {
      v8f c = v8f{};
      const _Float16* krow = &Kt[cur][(j * 16 + ln) * 72];
      c = wmma_f16(qa0, load_bfrag(krow, g), c);
      c = wmma_f16(qa1, load_bfrag(krow + 32, g), c);
      float mk = mask[b * SEQ + nb + j * 16 + ln];
#pragma unroll
      for (int r = 0; r < 8; ++r) sc[j][r] = c[r] * 0.125f + mk;
    }

    // online softmax (row = 8*g + r, col = ln)
    float p0[8], p1[8];
#pragma unroll
    for (int r = 0; r < 8; ++r) {
      float t = fmaxf(sc[0][r], sc[1][r]);
#pragma unroll
      for (int off = 1; off < 16; off <<= 1) t = fmaxf(t, __shfl_xor(t, off, 16));
      float mn  = fmaxf(mrun[r], t);
      float fac = __expf(mrun[r] - mn);
      mrun[r] = mn;
      p0[r] = __expf(sc[0][r] - mn);
      p1[r] = __expf(sc[1][r] - mn);
      float rs = p0[r] + p1[r];
#pragma unroll
      for (int off = 1; off < 16; off <<= 1) rs += __shfl_xor(rs, off, 16);
      lrun[r] = lrun[r] * fac + rs;
#pragma unroll
      for (int dt = 0; dt < 4; ++dt) o[dt][r] *= fac;
    }

    // P: C-layout -> per-wave LDS row-major, reload as A-fragment
    _Float16* pw = Pb[wave];
#pragma unroll
    for (int r = 0; r < 8; ++r) {
      int m = 8 * g + r;
      pw[m * 40 + ln]      = (_Float16)p0[r];
      pw[m * 40 + 16 + ln] = (_Float16)p1[r];
    }
    asm volatile("s_wait_dscnt 0" ::: "memory");  // same-wave LDS RAW fence
    v16h pa = load_afrag(&pw[ln * 40], g);

    // O += P @ V
#pragma unroll
    for (int dt = 0; dt < 4; ++dt)
      o[dt] = wmma_f16(pa, load_bfrag(&Vt[cur][(dt * 16 + ln) * 40], g), o[dt]);

    if (more) store_v(cur ^ 1, vnext);  // finish next-tile V staging
    wait_async0();
    __syncthreads();
    cur ^= 1;
  }

  // epilogue: normalize, store fp32 to [b][s][h*64+d]
  float inv[8];
#pragma unroll
  for (int r = 0; r < 8; ++r) inv[r] = 1.0f / lrun[r];
#pragma unroll
  for (int dt = 0; dt < 4; ++dt) {
#pragma unroll
    for (int r = 0; r < 8; ++r) {
      int m = 8 * g + r;
      int s = s0 + wave * 16 + m;
      int col = h * 64 + dt * 16 + ln;
      out[(size_t)(b * SEQ + s) * HIDDEN + col] = o[dt][r] * inv[r];
    }
  }
}

// ============================================================================
extern "C" void kernel_launch(void* const* d_in, const int* in_sizes, int n_in,
                              void* d_out, int out_size, void* d_ws,
                              size_t ws_size, hipStream_t stream) {
  const float* X    = (const float*)d_in[0];
  const float* mask = (const float*)d_in[1];
  const float* Wq   = (const float*)d_in[2];
  const float* bq   = (const float*)d_in[3];
  const float* Wk   = (const float*)d_in[4];
  const float* bk   = (const float*)d_in[5];
  const float* Wv   = (const float*)d_in[6];
  const float* bv   = (const float*)d_in[7];
  float* out = (float*)d_out;

  const size_t qkv_elems = (size_t)BATCH * NUM_HEADS * SEQ * HEAD_DIM;  // 8.4M
  const size_t x_elems   = (size_t)BATCH * SEQ * HIDDEN;                // 8.4M
  const size_t w_elems   = (size_t)HIDDEN * HIDDEN;                     // 1.05M

  _Float16* ws  = (_Float16*)d_ws;
  _Float16* qkv = ws;                      // 3 * 8.4M f16   (48 MB)
  _Float16* Xh  = ws + 3 * qkv_elems;      // 8.4M f16       (16 MB)
  _Float16* Wh  = Xh + x_elems;            // 3 * 1.05M f16  ( 6 MB)

  // fp32 -> f16 pre-pass (pure HBM streaming)
  f32_to_f16_kernel<<<(int)(x_elems / 8 / 256), 256, 0, stream>>>(X, Xh,
                                                                  (int)x_elems);
  f32_to_f16_kernel<<<(int)(w_elems / 8 / 256), 256, 0, stream>>>(
      Wq, Wh + 0 * w_elems, (int)w_elems);
  f32_to_f16_kernel<<<(int)(w_elems / 8 / 256), 256, 0, stream>>>(
      Wk, Wh + 1 * w_elems, (int)w_elems);
  f32_to_f16_kernel<<<(int)(w_elems / 8 / 256), 256, 0, stream>>>(
      Wv, Wh + 2 * w_elems, (int)w_elems);

  qkv_proj_kernel<<<dim3((BATCH * SEQ) / 128, HIDDEN / 128, 3), 256, 0,
                    stream>>>(Xh, Wh, bq, bk, bv, qkv);

  attention_kernel<<<dim3(BATCH * NUM_HEADS, SEQ / 128), 256, 0, stream>>>(
      qkv, qkv + qkv_elems, qkv + 2 * qkv_elems, mask, out);
}
